// MyViT_26164940767777
// MI455X (gfx1250) — compile-verified, matrix operand order checked by reading, not run on hardware
//
#include <hip/hip_runtime.h>
#include <math.h>

// ---------------------------------------------------------------------------
// ViT-Base/16 forward, MI455X (gfx1250):
//  - bf16 WMMA (v_wmma_f32_16x16x32_bf16) with f32 accumulation
//  - weights pre-packed to bf16 in WMMA-fragment tile layout, streamed to LDS
//    with gfx1250 async global->LDS copies (ASYNCcnt), no VGPR round-trip
//  - activations converted fp32->bf16 during LDS staging (b128 loads/stores)
// ---------------------------------------------------------------------------

typedef __attribute__((ext_vector_type(16))) __bf16 v16bf;
typedef __attribute__((ext_vector_type(8)))  __bf16 v8bf;
typedef __attribute__((ext_vector_type(8)))  float  v8f;

__device__ __forceinline__ unsigned short bfbits(float f) {
    union { float f; unsigned u; } v; v.f = f;
    unsigned r = v.u + 0x7FFFu + ((v.u >> 16) & 1u);   // round-to-nearest-even
    return (unsigned short)(r >> 16);
}
__device__ __forceinline__ __bf16 f2bf(float f) {
    unsigned short h = bfbits(f);
    __bf16 b; __builtin_memcpy(&b, &h, 2); return b;
}
__device__ __forceinline__ unsigned pk2(float a, float b) {
    return (unsigned)bfbits(a) | ((unsigned)bfbits(b) << 16);
}
// K-column permutation within a 32-chunk (involution): groups of 8 reordered
// 0,2,1,3 so a lane-half's 16 fragment elements are contiguous in LDS.
__device__ __forceinline__ int permc(int k) {
    int g = k >> 3; int ng = ((g & 1) << 1) | (g >> 1);
    return (ng << 3) | (k & 7);
}
__device__ __forceinline__ int imin(int a, int b) { return a < b ? a : b; }

// Packed-weight tile geometry: [nb][kc][128 n][40 perm-k] bf16 (10240 B/tile)
#define PKT_ELE (128 * 40)
#define PKT_BYT (128 * 40 * 2)

// ---------------------------------------------------------------------------
// Pack fp32 weights [K x N] (row-major, leading dim ldb) into swizzled bf16
// tiles, zero-padded to multiples of 128 (N) / 32 (K). Batched over z.
// ---------------------------------------------------------------------------
__global__ void pack_weights(const float* __restrict__ W, unsigned short* __restrict__ out,
                             int K, int N, long long ldb, long long sIn, long long sOut)
{
    const int KC = (K + 31) >> 5, NBk = (N + 127) >> 7;
    const long long total = (long long)NBk * KC * PKT_ELE;
    long long idx = (long long)blockIdx.x * 256 + threadIdx.x;
    if (idx >= total) return;
    const int z  = blockIdx.z;
    int pk = (int)(idx % 40);
    long long t = idx / 40;
    int nl = (int)(t % 128); t /= 128;
    int kc = (int)(t % KC);
    int nb = (int)(t / KC);
    float v = 0.0f;
    if (pk < 32) {
        int k = kc * 32 + permc(pk);     // permc is an involution
        int n = nb * 128 + nl;
        if (k < K && n < N) v = W[z * sIn + (long long)k * ldb + n];
    }
    out[z * sOut + idx] = bfbits(v);
}

// ---------------------------------------------------------------------------
// Batched GEMM: C = epilogue(alpha * A*B + bias).
//  A: fp32 row-major [M x K] (converted to bf16 while staging).
//  B: PB=1 -> packed bf16 tiles (async global->LDS copy);
//     PB=0 -> fp32 [K x N] (or [N x K] when TRANSB=1).
// Block tile 64x128, 8 waves each 32x32, double-buffered LDS.
// blockIdx.z = zn*Hdim + zh with independent strides for A/B/C/bias.
// EPI: 0 = store (+bias), 1 = GELU(+bias) store, 2 = residual += (+bias)
// ---------------------------------------------------------------------------
template <int TRANSB, int PB, int EPI>
__global__ __launch_bounds__(256) void gemm_kernel(
    const float* __restrict__ A, const void* __restrict__ B,
    const float* __restrict__ bias, float* __restrict__ C,
    int M, int N, int K,
    long long lda, long long ldb, long long ldc,
    long long sAn, long long sAh, long long sBn, long long sBh,
    long long sCn, long long sCh, long long sBiasH,
    int Hdim, float alpha)
{
    __shared__ __bf16 As[2][64][40];    // [buf][m][perm-k], row 80B (16B aligned)
    __shared__ __bf16 Bs[2][128][40];   // [buf][n][perm-k]

    const int tid    = threadIdx.x;
    const int lane   = tid & 31;
    const int wid    = tid >> 5;
    const int wm     = wid >> 2;          // 0..1
    const int wn     = wid & 3;           // 0..3
    const int mrow   = lane & 15;
    const int half16 = lane >> 4;         // 0/1 -> which 16-element half
    const int m0     = blockIdx.y * 64;
    const int n0     = blockIdx.x * 128;
    const int zn     = blockIdx.z / Hdim;
    const int zh     = blockIdx.z % Hdim;

    const float* Ap = A + (long long)zn * sAn + (long long)zh * sAh;
    const long long offBe = (long long)zn * sBn + (long long)zh * sBh;
    const float* Bf = (const float*)B + offBe;        // fp32 B (PB=0)
    const char*  Bq = (const char*)B + offBe * 2;     // packed bf16 B (PB=1)
    const long long offC = (long long)zn * sCn + (long long)zh * sCh;

    auto stageA = [&](int buf, int k0) {
        if ((m0 + 64 <= M) && (k0 + 32 <= K) && ((lda & 3) == 0)) {
            // one perm-group of 8 K per thread: 2x global b128 -> 1x ds b128
            int r = tid >> 2, c8 = (tid & 3) << 3;
            const float* s = Ap + (long long)(m0 + r) * lda + k0 + c8;
            const float4 f0 = *(const float4*)(s);
            const float4 f1 = *(const float4*)(s + 4);
            uint4 u;
            u.x = pk2(f0.x, f0.y); u.y = pk2(f0.z, f0.w);
            u.z = pk2(f1.x, f1.y); u.w = pk2(f1.z, f1.w);
            *(uint4*)&As[buf][r][permc(c8)] = u;       // ds_store_b128
        } else {
            #pragma unroll
            for (int i = 0; i < 8; ++i) {
                int idx = tid + 256 * i;
                int r = idx >> 5, c = idx & 31;
                int gm = m0 + r, gk = k0 + c;
                // branchless: clamp address (always legal), select 0 if OOB
                float v = Ap[(long long)imin(gm, M - 1) * lda + imin(gk, K - 1)];
                v = (gm < M && gk < K) ? v : 0.0f;
                As[buf][r][permc(c)] = f2bf(v);
            }
        }
    };

    auto stageB = [&](int buf, int k0) {
        if constexpr (PB) {
            // Packed bf16 tile (10240B) -> async global->LDS copy, ASYNCcnt.
            // Exact coverage, no guards: 512 x b128 + 256 x b64 per block.
            const long long blk = ((long long)(n0 >> 7) * ((K + 31) >> 5) + (k0 >> 5)) * PKT_BYT;
            const char* src = Bq + blk;
            unsigned ldsb = (unsigned)(unsigned long long)(void*)&Bs[buf][0][0];
            #pragma unroll
            for (int i = 0; i < 2; ++i) {
                int off = (tid + 256 * i) * 16;
                unsigned long long ga = (unsigned long long)(const void*)(src + off);
                unsigned la = ldsb + off;
                asm volatile("global_load_async_to_lds_b128 %0, %1, off"
                             :: "v"(la), "v"(ga) : "memory");
            }
            {
                int off = 8192 + tid * 8;
                unsigned long long ga = (unsigned long long)(const void*)(src + off);
                unsigned la = ldsb + off;
                asm volatile("global_load_async_to_lds_b64 %0, %1, off"
                             :: "v"(la), "v"(ga) : "memory");
            }
        } else if constexpr (TRANSB) {
            // B stored [N x K]: contiguous along K -> b128 loads, b128 stores
            if ((n0 + 128 <= N) && (k0 + 32 <= K) && ((ldb & 3) == 0)) {
                #pragma unroll
                for (int i = 0; i < 2; ++i) {
                    int r = tid >> 1, c8 = ((tid & 1) << 4) + (i << 3);  // r = n
                    const float* s = Bf + (long long)(n0 + r) * ldb + k0 + c8;
                    const float4 f0 = *(const float4*)(s);
                    const float4 f1 = *(const float4*)(s + 4);
                    uint4 u;
                    u.x = pk2(f0.x, f0.y); u.y = pk2(f0.z, f0.w);
                    u.z = pk2(f1.x, f1.y); u.w = pk2(f1.z, f1.w);
                    *(uint4*)&Bs[buf][r][permc(c8)] = u;
                }
            } else {
                #pragma unroll
                for (int i = 0; i < 16; ++i) {
                    int idx = tid + 256 * i;
                    int r = idx >> 5, c = idx & 31;         // r = n, c = k
                    int gn = n0 + r, gk = k0 + c;
                    float v = Bf[(long long)imin(gn, N - 1) * ldb + imin(gk, K - 1)];
                    v = (gn < N && gk < K) ? v : 0.0f;
                    Bs[buf][r][permc(c)] = f2bf(v);
                }
            }
        } else {
            // B stored [K x N]: coalesced float4 along N, scatter b16 to [n][perm-k]
            if ((n0 + 128 <= N) && (k0 + 32 <= K) && ((ldb & 3) == 0)) {
                #pragma unroll
                for (int i = 0; i < 4; ++i) {
                    int idx = tid + 256 * i;
                    int r = idx >> 5, c4 = (idx & 31) << 2;  // r = k 0..31, c4 = n
                    const float4 f = *(const float4*)(Bf + (long long)(k0 + r) * ldb + n0 + c4);
                    int pr = permc(r);
                    Bs[buf][c4 + 0][pr] = f2bf(f.x);
                    Bs[buf][c4 + 1][pr] = f2bf(f.y);
                    Bs[buf][c4 + 2][pr] = f2bf(f.z);
                    Bs[buf][c4 + 3][pr] = f2bf(f.w);
                }
            } else {
                #pragma unroll
                for (int i = 0; i < 16; ++i) {
                    int idx = tid + 256 * i;
                    int r = idx >> 7, c = idx & 127;         // r = k, c = n
                    int gk = k0 + r, gn = n0 + c;
                    float v = Bf[(long long)imin(gk, K - 1) * ldb + imin(gn, N - 1)];
                    v = (gk < K && gn < N) ? v : 0.0f;
                    Bs[buf][c][permc(r)] = f2bf(v);
                }
            }
        }
    };

    v8f acc[2][2] = {};
    const int fbase = half16 << 4;

    auto compute = [&](int buf) {
        v16bf af[2], bv[2];
        #pragma unroll
        for (int mi = 0; mi < 2; ++mi) {
            const v8bf* p = (const v8bf*)&As[buf][wm * 32 + mi * 16 + mrow][fbase];
            af[mi] = __builtin_shufflevector(p[0], p[1],
                     0,1,2,3,4,5,6,7,8,9,10,11,12,13,14,15);   // 2x ds_load_b128
        }
        #pragma unroll
        for (int ni = 0; ni < 2; ++ni) {
            const v8bf* p = (const v8bf*)&Bs[buf][wn * 32 + ni * 16 + mrow][fbase];
            bv[ni] = __builtin_shufflevector(p[0], p[1],
                     0,1,2,3,4,5,6,7,8,9,10,11,12,13,14,15);
        }
        #pragma unroll
        for (int mi = 0; mi < 2; ++mi)
            #pragma unroll
            for (int ni = 0; ni < 2; ++ni)
                acc[mi][ni] = __builtin_amdgcn_wmma_f32_16x16x32_bf16(
                    false, af[mi], false, bv[ni], (short)0, acc[mi][ni], false, false);
    };

    const int nk = (K + 31) >> 5;
    stageA(0, 0);
    stageB(0, 0);
    if constexpr (PB) asm volatile("s_wait_asynccnt 0" ::: "memory");
    __syncthreads();
    for (int kc = 0; kc < nk; ++kc) {
        const int cur = kc & 1;
        if (kc + 1 < nk) {                       // stage next chunk into other buffer
            stageA(cur ^ 1, (kc + 1) << 5);
            stageB(cur ^ 1, (kc + 1) << 5);
        }
        if (kc + 2 < nk)                         // speculative prefetch (chunk after next)
            __builtin_prefetch(Ap + (long long)(m0 + (tid & 63)) * lda + ((kc + 2) << 5), 0, 1);
        compute(cur);
        if constexpr (PB) asm volatile("s_wait_asynccnt 0" ::: "memory");
        __syncthreads();
    }

    // ---- epilogue ----
    const float* biasp = bias ? (bias + (long long)zh * sBiasH) : nullptr;
    #pragma unroll
    for (int mi = 0; mi < 2; ++mi)
        #pragma unroll
        for (int ni = 0; ni < 2; ++ni)
            #pragma unroll
            for (int r = 0; r < 8; ++r) {
                int gm = m0 + wm * 32 + mi * 16 + (half16 << 3) + r;
                int gn = n0 + wn * 32 + ni * 16 + mrow;
                if (gm < M && gn < N) {
                    float v = acc[mi][ni][r] * alpha;
                    if (biasp) v += biasp[gn];
                    if (EPI == 1) v = 0.5f * v * (1.0f + erff(v * 0.70710678118f));
                    long long ci = offC + (long long)gm * ldc + gn;
                    if (EPI == 2) C[ci] += v; else C[ci] = v;
                }
            }
}

// ---------------------------------------------------------------------------
// Patch extraction: x[N,3,224,224] -> patches[N*196, 768] (c-major 16x16)
// ---------------------------------------------------------------------------
__global__ void extract_patches(const float* __restrict__ x, float* __restrict__ p, long long total)
{
    long long idx = (long long)blockIdx.x * 256 + threadIdx.x;
    if (idx >= total) return;
    int col = (int)(idx % 768);
    long long t = idx / 768;
    int pp = (int)(t % 196);
    int n  = (int)(t / 196);
    int c  = col >> 8, rem = col & 255, iy = rem >> 4, ix = rem & 15;
    int py = pp / 14, px = pp % 14;
    p[idx] = x[(((long long)n * 3 + c) * 224 + py * 16 + iy) * 224 + px * 16 + ix];
}

// ---------------------------------------------------------------------------
// Assemble tokens: cls + embedded patches, plus sinusoidal positions.
// ---------------------------------------------------------------------------
__global__ void assemble_tokens(const float* __restrict__ tmp, const float* __restrict__ cls,
                                float* __restrict__ tokens, long long total)
{
    long long idx = (long long)blockIdx.x * 256 + threadIdx.x;
    if (idx >= total) return;
    int col = (int)(idx % 768);
    long long t = idx / 768;
    int s = (int)(t % 197);
    int n = (int)(t / 197);
    float ang = (float)s * powf(10000.0f, -2.0f * (float)col / 768.0f);
    float pos = ((col & 1) == 0) ? sinf(ang) : cosf(ang);
    float base = (s == 0) ? cls[col] : tmp[((long long)n * 196 + (s - 1)) * 768 + col];
    tokens[idx] = base + pos;
}

// ---------------------------------------------------------------------------
// LayerNorm over last dim (cols=768), one 256-thread block per row.
// ---------------------------------------------------------------------------
__global__ __launch_bounds__(256) void layernorm_kernel(
    const float* __restrict__ x, const float* __restrict__ g, const float* __restrict__ b,
    float* __restrict__ out, int cols)
{
    __shared__ float s1[256], s2[256];
    long long row = blockIdx.x;
    const float* xr = x + row * cols;
    float a = 0.f, q = 0.f;
    for (int c = threadIdx.x; c < cols; c += 256) { float v = xr[c]; a += v; q += v * v; }
    s1[threadIdx.x] = a; s2[threadIdx.x] = q; __syncthreads();
    for (int st = 128; st > 0; st >>= 1) {
        if ((int)threadIdx.x < st) { s1[threadIdx.x] += s1[threadIdx.x + st]; s2[threadIdx.x] += s2[threadIdx.x + st]; }
        __syncthreads();
    }
    float mean = s1[0] / cols;
    float var  = s2[0] / cols - mean * mean;
    float inv  = rsqrtf(var + 1e-5f);
    for (int c = threadIdx.x; c < cols; c += 256)
        out[row * cols + c] = (xr[c] - mean) * inv * g[c] + b[c];
}

// ---------------------------------------------------------------------------
// Row softmax: one wave32 per row (8 rows per 256-thread block).
// ---------------------------------------------------------------------------
__global__ __launch_bounds__(256) void softmax_rows(
    const float* __restrict__ in, float* __restrict__ out, long long rows, int cols)
{
    int wid = threadIdx.x >> 5, lane = threadIdx.x & 31;
    long long row = (long long)blockIdx.x * 8 + wid;
    if (row >= rows) return;
    const float* pi = in  + row * cols;
    float*       po = out + row * cols;
    float mx = -3.4e38f;
    for (int c = lane; c < cols; c += 32) mx = fmaxf(mx, pi[c]);
    for (int o = 16; o > 0; o >>= 1) mx = fmaxf(mx, __shfl_xor(mx, o, 32));
    float sum = 0.f;
    for (int c = lane; c < cols; c += 32) { float e = __expf(pi[c] - mx); po[c] = e; sum += e; }
    for (int o = 16; o > 0; o >>= 1) sum += __shfl_xor(sum, o, 32);
    float inv = 1.f / sum;
    for (int c = lane; c < cols; c += 32) po[c] *= inv;
}

// ---------------------------------------------------------------------------
template <int TRANSB, int PB, int EPI>
static inline void gemm(hipStream_t st, const float* A, const void* B, const float* bias, float* C,
                        int M, int N, int K, long long lda, long long ldb, long long ldc,
                        long long sAn, long long sAh, long long sBn, long long sBh,
                        long long sCn, long long sCh, long long sBiasH,
                        int Hdim, int batches, float alpha)
{
    dim3 g((N + 127) / 128, (M + 63) / 64, batches);
    gemm_kernel<TRANSB, PB, EPI><<<g, dim3(256), 0, st>>>(
        A, B, bias, C, M, N, K, lda, ldb, ldc,
        sAn, sAh, sBn, sBh, sCn, sCh, sBiasH, Hdim, alpha);
}

static inline void pack(hipStream_t st, const float* W, unsigned short* out,
                        int K, int N, long long ldb, long long sIn, long long sOut, int batches)
{
    int KC = (K + 31) >> 5, NBk = (N + 127) >> 7;
    long long total = (long long)NBk * KC * PKT_ELE;
    dim3 g((unsigned)((total + 255) / 256), 1, batches);
    pack_weights<<<g, dim3(256), 0, st>>>(W, out, K, N, ldb, sIn, sOut);
}

extern "C" void kernel_launch(void* const* d_in, const int* in_sizes, int n_in,
                              void* d_out, int out_size, void* d_ws, size_t ws_size,
                              hipStream_t stream)
{
    (void)in_sizes; (void)n_in; (void)out_size; (void)ws_size;
    const float* x     = (const float*)d_in[0];
    const float* Wemb  = (const float*)d_in[1];
    const float* bemb  = (const float*)d_in[2];
    const float* cls   = (const float*)d_in[3];
    const float* Wq    = (const float*)d_in[4];
    const float* bq    = (const float*)d_in[5];
    const float* Wk    = (const float*)d_in[6];
    const float* bk    = (const float*)d_in[7];
    const float* Wv    = (const float*)d_in[8];
    const float* bv    = (const float*)d_in[9];
    const float* ln1g  = (const float*)d_in[10];
    const float* ln1b  = (const float*)d_in[11];
    const float* ln2g  = (const float*)d_in[12];
    const float* ln2b  = (const float*)d_in[13];
    const float* W1    = (const float*)d_in[14];
    const float* b1    = (const float*)d_in[15];
    const float* W2    = (const float*)d_in[16];
    const float* b2    = (const float*)d_in[17];
    const float* Whead = (const float*)d_in[18];
    const float* bhead = (const float*)d_in[19];
    float* out = (float*)d_out;

    constexpr int NB = 32, S = 197, D = 768, NH = 12, DH = 64, DFF = 3072, NL = 12;
    constexpr long long Mtot = (long long)NB * S;       // 6304
    constexpr long long TOK  = (long long)NB * S * D;   // 4,841,472
    constexpr long long R2   = 19365888;                // embed tmp | scores | mlp hidden

    // Workspace layout (floats then packed bf16), lifetime-aliased:
    float* ws     = (float*)d_ws;
    float* tokens = ws;                 // TOK
    float* qb     = tokens + TOK;       // [h][m][64] = TOK
    float* kb     = qb + TOK;
    float* vb     = kb + TOK;
    float* r1     = vb + TOK;           // patches | LN output  (TOK)
    float* r2     = r1 + TOK;           // embed tmp | scores | mlp hidden (R2)
    unsigned short* pW1  = (unsigned short*)(r2 + R2);  // 24*24*PKT_ELE (also Wemb/Whead)
    unsigned short* pW2  = pW1 + (long long)24 * 24 * PKT_ELE;  // 6*96*PKT_ELE
    unsigned short* pQ   = pW2 + (long long)6 * 96 * PKT_ELE;   // NH * 2*PKT_ELE each
    unsigned short* pK   = pQ + (long long)NH * 2 * PKT_ELE;
    unsigned short* pV   = pK + (long long)NH * 2 * PKT_ELE;

    // ---- patch embedding ----
    {
        long long tp = (long long)NB * 196 * D;
        extract_patches<<<dim3((unsigned)((tp + 255) / 256)), dim3(256), 0, stream>>>(x, r1, tp);
        pack(stream, Wemb, pW1, D, D, D, 0, 0, 1);
        gemm<0, 1, 0>(stream, r1, pW1, bemb, r2, NB * 196, D, D, D, 0, D,
                      0, 0, 0, 0, 0, 0, 0, 1, 1, 1.0f);
        assemble_tokens<<<dim3((unsigned)((TOK + 255) / 256)), dim3(256), 0, stream>>>(r2, cls, tokens, TOK);
    }

    for (int l = 0; l < NL; ++l) {
        const long long wq_off = (long long)l * NH * DH * DH;
        const long long bq_off = (long long)l * NH * DH;

        // pack this layer's weights to swizzled bf16
        pack(stream, Wq + wq_off, pQ, DH, DH, DH, (long long)DH * DH, 2 * PKT_ELE, NH);
        pack(stream, Wk + wq_off, pK, DH, DH, DH, (long long)DH * DH, 2 * PKT_ELE, NH);
        pack(stream, Wv + wq_off, pV, DH, DH, DH, (long long)DH * DH, 2 * PKT_ELE, NH);
        pack(stream, W1 + (long long)l * D * DFF, pW1, D, DFF, DFF, 0, 0, 1);
        pack(stream, W2 + (long long)l * DFF * D, pW2, DFF, D, D, 0, 0, 1);

        // LN1 -> r1
        layernorm_kernel<<<dim3((unsigned)Mtot), dim3(256), 0, stream>>>(
            tokens, ln1g + (long long)l * D, ln1b + (long long)l * D, r1, D);

        // Q/K/V per-head projection (batched over heads): [h][m][64]
        gemm<0, 1, 0>(stream, r1, pQ, bq + bq_off, qb, (int)Mtot, DH, DH,
                      D, 0, DH, 0, DH, 0, 2 * PKT_ELE, 0, Mtot * DH, DH, NH, NH, 1.0f);
        gemm<0, 1, 0>(stream, r1, pK, bk + bq_off, kb, (int)Mtot, DH, DH,
                      D, 0, DH, 0, DH, 0, 2 * PKT_ELE, 0, Mtot * DH, DH, NH, NH, 1.0f);
        gemm<0, 1, 0>(stream, r1, pV, bv + bq_off, vb, (int)Mtot, DH, DH,
                      D, 0, DH, 0, DH, 0, 2 * PKT_ELE, 0, Mtot * DH, DH, NH, NH, 1.0f);

        // scores[n][h] = (Q Kt) * 1/sqrt(dh)  (batched over n*h, B transposed, fp32 B)
        gemm<1, 0, 0>(stream, qb, kb, nullptr, r2, S, S, DH, DH, DH, S,
                      (long long)S * DH, Mtot * DH, (long long)S * DH, Mtot * DH,
                      (long long)NH * S * S, (long long)S * S, 0, NH, NB * NH, 0.125f);

        // softmax over keys
        {
            long long rows = (long long)NB * NH * S;
            softmax_rows<<<dim3((unsigned)((rows + 7) / 8)), dim3(256), 0, stream>>>(r2, r2, rows, S);
        }

        // attn output: tokens[n][s][h*64+e] += P V  (residual epilogue, fp32 B)
        gemm<0, 0, 2>(stream, r2, vb, nullptr, tokens, S, DH, S, S, DH, D,
                      (long long)NH * S * S, (long long)S * S, (long long)S * DH, Mtot * DH,
                      (long long)S * D, DH, 0, NH, NB * NH, 1.0f);

        // LN2 -> r1
        layernorm_kernel<<<dim3((unsigned)Mtot), dim3(256), 0, stream>>>(
            tokens, ln2g + (long long)l * D, ln2b + (long long)l * D, r1, D);

        // MLP: r2 = gelu(r1 @ W1 + b1); tokens += r2 @ W2 + b2 (packed weights)
        gemm<0, 1, 1>(stream, r1, pW1, b1 + (long long)l * DFF, r2,
                      (int)Mtot, DFF, D, D, 0, DFF, 0, 0, 0, 0, 0, 0, 0, 1, 1, 1.0f);
        gemm<0, 1, 2>(stream, r2, pW2, b2 + (long long)l * D, tokens,
                      (int)Mtot, D, DFF, DFF, 0, D, 0, 0, 0, 0, 0, 0, 0, 1, 1, 1.0f);
    }

    // classifier head on cls token rows (row n at tokens + n*S*D) + softmax
    pack(stream, Whead, pW1, D, 1000, 1000, 0, 0, 1);
    gemm<0, 1, 0>(stream, tokens, pW1, bhead, out, NB, 1000, D,
                  (long long)S * D, 0, 1000, 0, 0, 0, 0, 0, 0, 0, 1, 1, 1.0f);
    softmax_rows<<<dim3((32 + 7) / 8), dim3(256), 0, stream>>>(out, out, NB, 1000);
}
